// infusion_KG_QA_3_86938728006070
// MI455X (gfx1250) — matrix operand order repaired and checked by reading, not compiled, not run on hardware
//
#include <hip/hip_runtime.h>
#include <hip/hip_bf16.h>

#define B_    8
#define CLEN  2048
#define QLEN  512
#define DDIM  1024
#define XDIM  4096

typedef __bf16 bf16_t;
typedef bf16_t v16bf __attribute__((ext_vector_type(16)));
typedef bf16_t v8bf  __attribute__((ext_vector_type(8)));
typedef float  v8f   __attribute__((ext_vector_type(8)));

__device__ __forceinline__ unsigned short f2bf(float f) {
  union { float f; unsigned int u; } v; v.f = f;
  unsigned int u = v.u;
  unsigned int r = (u + 0x7FFFu + ((u >> 16) & 1u)) >> 16;  // round-to-nearest-even
  return (unsigned short)r;
}

// Async copy 16 bytes global -> LDS (gfx1250, ASYNCcnt-tracked, no VGPR staging).
// lds_off: wave-relative LDS byte address (low 32 bits of a generic shared pointer).
__device__ __forceinline__ void async_g2l_b128(unsigned lds_off, const void* gaddr) {
  asm volatile("global_load_async_to_lds_b128 %0, %1, off"
               :: "v"(lds_off), "v"(gaddr) : "memory");
}
__device__ __forceinline__ void wait_asynccnt0() {
  asm volatile("s_wait_asynccnt 0x0" ::: "memory");
}

// A-matrix 16x32 bf16 fragment (ISA 7.12.2): lane M = lane&15;
// lanes 0-15 hold K = {k0..k0+7, k0+16..k0+23}; lanes 16-31 hold K = {k0+8..k0+15, k0+24..k0+31}.
__device__ __forceinline__ v16bf load_a(const unsigned short* base, size_t ld, int k0, int lane) {
  const unsigned short* p = base + (size_t)(lane & 15) * ld;
  int off = k0 + ((lane >> 4) << 3);
  v8bf lo = *(const v8bf*)(p + off);
  v8bf hi = *(const v8bf*)(p + off + 16);
  return __builtin_shufflevector(lo, hi, 0,1,2,3,4,5,6,7,8,9,10,11,12,13,14,15);
}

// B-matrix 32x16 bf16 fragment: lane column N = lane&15; lanes 0-15 hold K={k0..k0+7, k0+16..k0+23},
// lanes 16-31 hold K={k0+8..k0+15, k0+24..k0+31}. `base` points at row n0 of an [N,K]-major matrix.
__device__ __forceinline__ v16bf load_b(const unsigned short* base, size_t ld, int k0, int lane) {
  const unsigned short* p = base + (size_t)(lane & 15) * ld;
  int off = k0 + ((lane >> 4) << 3);
  v8bf lo = *(const v8bf*)(p + off);
  v8bf hi = *(const v8bf*)(p + off + 16);
  return __builtin_shufflevector(lo, hi, 0,1,2,3,4,5,6,7,8,9,10,11,12,13,14,15);
}

// ---------------- prep kernels ----------------

__global__ void prep_q_kernel(const float* __restrict__ q, const float* __restrict__ w_q,
                              const float* __restrict__ b_q,
                              unsigned short* __restrict__ q_bf,
                              unsigned short* __restrict__ q_t_bf,
                              float* __restrict__ rq) {
  size_t row = blockIdx.x;                 // b*QLEN + qi
  int b = (int)(row / QLEN), qi = (int)(row % QLEN);
  const float* src = q + row * DDIM;
  __shared__ float red[256];
  float acc = 0.f;
  for (int d = threadIdx.x; d < DDIM; d += 256) {
    float v = src[d];
    unsigned short h = f2bf(v);
    q_bf[row * DDIM + d] = h;
    q_t_bf[((size_t)b * DDIM + d) * QLEN + qi] = h;   // transpose: [B][D][Q]
    acc += v * w_q[d];
  }
  red[threadIdx.x] = acc; __syncthreads();
  for (int s = 128; s > 0; s >>= 1) {
    if (threadIdx.x < (unsigned)s) red[threadIdx.x] += red[threadIdx.x + s];
    __syncthreads();
  }
  if (threadIdx.x == 0) rq[row] = red[0] + b_q[0];
}

__global__ void prep_c_kernel(const float* __restrict__ c, const float* __restrict__ w_cq,
                              const float* __restrict__ w_c, const float* __restrict__ b_c,
                              unsigned short* __restrict__ cw_bf, float* __restrict__ rc) {
  size_t row = blockIdx.x;                 // b*CLEN + ci
  const float* src = c + row * DDIM;
  __shared__ float red[256];
  float acc = 0.f;
  for (int d = threadIdx.x; d < DDIM; d += 256) {
    float v = src[d];
    cw_bf[row * DDIM + d] = f2bf(v * w_cq[d]);
    acc += v * w_c[d];
  }
  red[threadIdx.x] = acc; __syncthreads();
  for (int s = 128; s > 0; s >>= 1) {
    if (threadIdx.x < (unsigned)s) red[threadIdx.x] += red[threadIdx.x + s];
    __syncthreads();
  }
  if (threadIdx.x == 0) rc[row] = red[0] + b_c[0];
}

__global__ void cvt_bf16_kernel(const float* __restrict__ src, unsigned short* __restrict__ dst, int n) {
  int i = blockIdx.x * blockDim.x + threadIdx.x;
  if (i < n) dst[i] = f2bf(src[i]);
}

// ---------------- fused attention: S = cw@q^T (+bias), softmax, c2q = A@q ----------------
// block = 64 threads = 2 waves; each wave owns 16 context rows.
// dyn LDS per wave: A-tile 16x1024 bf16 (32KB, async-staged; later reused for bf16 probs)
//                 + scores 16x512 f32 (32KB).  Total 128KB/block.

__global__ void attn_kernel(const unsigned short* __restrict__ cw_bf,
                            const unsigned short* __restrict__ q_bf,
                            const unsigned short* __restrict__ q_t_bf,
                            const float* __restrict__ rc, const float* __restrict__ rq,
                            const float* __restrict__ b_cq_p,
                            const float* __restrict__ c,
                            float* __restrict__ mrow,
                            unsigned short* __restrict__ X) {
  extern __shared__ char smem[];
  const int wave = threadIdx.x >> 5;
  const int lane = threadIdx.x & 31;
  const int b = blockIdx.y;
  const int c0 = blockIdx.x * 32 + wave * 16;

  unsigned short* s_a = (unsigned short*)smem + (size_t)wave * 16 * DDIM;       // A tile / probs
  float* s_s = (float*)(smem + (size_t)2 * 16 * DDIM * 2) + (size_t)wave * 16 * QLEN;
  unsigned short* s_p = s_a;                                                    // probs alias
  const float b_cq = b_cq_p[0];

  const unsigned short* Abase = cw_bf + ((size_t)b * CLEN + c0) * DDIM;
  const unsigned short* Qbase = q_bf + (size_t)b * QLEN * DDIM;

  // ---- phase 0: async-stage the 16x1024 bf16 A tile into LDS (contiguous copy) ----
  {
    unsigned lbase = (unsigned)(size_t)s_a;
    for (int t = lane; t < (16 * DDIM) / 8; t += 32)
      async_g2l_b128(lbase + (unsigned)t * 16u, Abase + (size_t)t * 8);
    wait_asynccnt0();      // wave-local tile: no cross-wave barrier needed
  }

  // preload per-lane rc values (8 rows this lane's D-fragment touches)
  const int rbase = (lane >> 4) << 3;
  float rcv[8];
#pragma unroll
  for (int v = 0; v < 8; ++v) rcv[v] = rc[(size_t)b * CLEN + c0 + v + rbase];

  // ---- phase 1: scores, 16x512 tile via WMMA over K=1024, A from LDS ----
#pragma unroll 1
  for (int nt = 0; nt < QLEN / 16; ++nt) {
    const int n0 = nt * 16;
    v8f acc = {};
#pragma unroll 4
    for (int k0 = 0; k0 < DDIM; k0 += 32) {
      v16bf a  = load_a(s_a, DDIM, k0, lane);                       // ds_load
      v16bf bm = load_b(Qbase + (size_t)n0 * DDIM, DDIM, k0, lane); // global
      acc = __builtin_amdgcn_wmma_f32_16x16x32_bf16(false, a, false, bm, (short)0, acc, false, false);
    }
    const int col = n0 + (lane & 15);
    const float rqv = rq[(size_t)b * QLEN + col] + b_cq;
#pragma unroll
    for (int v = 0; v < 8; ++v)
      s_s[(v + rbase) * QLEN + col] = acc[v] + rcv[v] + rqv;
  }

  __syncthreads();

  // ---- phase 2: wave-local softmax over Q=512 per row; record row max; probs -> LDS ----
  for (int r = 0; r < 16; ++r) {
    float vals[16];
    float mx = -3.4e38f;
#pragma unroll
    for (int j = 0; j < 16; ++j) {
      vals[j] = s_s[r * QLEN + lane + 32 * j];
      mx = fmaxf(mx, vals[j]);
    }
    for (int off = 16; off > 0; off >>= 1) mx = fmaxf(mx, __shfl_xor(mx, off, 32));
    float sum = 0.f;
#pragma unroll
    for (int j = 0; j < 16; ++j) { vals[j] = __expf(vals[j] - mx); sum += vals[j]; }
    for (int off = 16; off > 0; off >>= 1) sum += __shfl_xor(sum, off, 32);
    const float inv = 1.0f / sum;
    if (lane == 0) mrow[(size_t)b * CLEN + c0 + r] = mx;   // max BEFORE softmax (for b_att)
#pragma unroll
    for (int j = 0; j < 16; ++j) s_p[r * QLEN + lane + 32 * j] = f2bf(vals[j] * inv);
  }

  __syncthreads();

  // ---- phase 3: c2q = A @ q via WMMA (A probs from LDS, B from q^T), write X slices 1 & 2 ----
  const unsigned short* Qt = q_t_bf + (size_t)b * DDIM * QLEN;
#pragma unroll 1
  for (int nt = 0; nt < DDIM / 16; ++nt) {
    const int n0 = nt * 16;
    v8f acc = {};
#pragma unroll 4
    for (int k0 = 0; k0 < QLEN; k0 += 32) {
      v16bf a  = load_a(s_p, QLEN, k0, lane);                        // ds_load
      v16bf bm = load_b(Qt + (size_t)n0 * QLEN, QLEN, k0, lane);
      acc = __builtin_amdgcn_wmma_f32_16x16x32_bf16(false, a, false, bm, (short)0, acc, false, false);
    }
    const int col = n0 + (lane & 15);
#pragma unroll
    for (int v = 0; v < 8; ++v) {
      size_t crow = (size_t)b * CLEN + c0 + v + rbase;
      float c2q = acc[v];
      float cv = c[crow * DDIM + col];
      unsigned short* xp = X + crow * XDIM;
      xp[DDIM + col]     = f2bf(c2q);                      // slice 1: c2q
      xp[2 * DDIM + col] = f2bf(cv * c2q);                 // slice 2: c * c2q
    }
  }
}

// ---------------- b_att = softmax over C of row maxes ----------------

__global__ void batt_kernel(const float* __restrict__ mrow, float* __restrict__ batt) {
  int b = blockIdx.x;
  __shared__ float red[256];
  const float* mp = mrow + (size_t)b * CLEN;
  float mx = -3.4e38f;
  for (int i = threadIdx.x; i < CLEN; i += 256) mx = fmaxf(mx, mp[i]);
  red[threadIdx.x] = mx; __syncthreads();
  for (int s = 128; s > 0; s >>= 1) {
    if (threadIdx.x < (unsigned)s) red[threadIdx.x] = fmaxf(red[threadIdx.x], red[threadIdx.x + s]);
    __syncthreads();
  }
  mx = red[0]; __syncthreads();
  float sum = 0.f;
  for (int i = threadIdx.x; i < CLEN; i += 256) sum += __expf(mp[i] - mx);
  red[threadIdx.x] = sum; __syncthreads();
  for (int s = 128; s > 0; s >>= 1) {
    if (threadIdx.x < (unsigned)s) red[threadIdx.x] += red[threadIdx.x + s];
    __syncthreads();
  }
  float inv = 1.0f / red[0];
  for (int i = threadIdx.x; i < CLEN; i += 256) batt[(size_t)b * CLEN + i] = __expf(mp[i] - mx) * inv;
}

// ---------------- q2c[b,d] = sum_c b_att[b,c] * c[b,c,d] ----------------

__global__ void q2c_kernel(const float* __restrict__ batt, const float* __restrict__ c,
                           float* __restrict__ q2c) {
  int b = blockIdx.y;
  int d = blockIdx.x * 256 + threadIdx.x;
  const float* cp = c + (size_t)b * CLEN * DDIM + d;
  const float* bp = batt + (size_t)b * CLEN;
  float acc = 0.f;
  for (int cc = 0; cc < CLEN; ++cc) acc += bp[cc] * cp[(size_t)cc * DDIM];
  q2c[(size_t)b * DDIM + d] = acc;
}

// ---------------- X slices 0 (c) and 3 (c * q2c) ----------------

__global__ void fillx_kernel(const float* __restrict__ c, const float* __restrict__ q2c,
                             unsigned short* __restrict__ X) {
  size_t row = blockIdx.x;                 // b*CLEN + ci
  int b = (int)(row / CLEN);
  const float* cp = c + row * DDIM;
  const float* qp = q2c + (size_t)b * DDIM;
  unsigned short* xp = X + row * XDIM;
  for (int d = threadIdx.x; d < DDIM; d += 256) {
    float cv = cp[d];
    xp[d] = f2bf(cv);
    xp[3 * DDIM + d] = f2bf(cv * qp[d]);
  }
}

// ---------------- Z = X @ W_l^T + b_l : [16384 x 4096] x [4096 x 1024] ----------------
// block = 256 threads = 8 waves. Block async-stages its 16x4096 bf16 A tile (128KB) in LDS
// once; all 8 waves consume it (8x global-traffic cut), B streamed from W_l (L2-resident).

__global__ void zgemm_kernel(const unsigned short* __restrict__ X,
                             const unsigned short* __restrict__ Wl_bf,
                             const float* __restrict__ b_l,
                             float* __restrict__ Z) {
  extern __shared__ char smem[];
  unsigned short* s_a = (unsigned short*)smem;             // 16 x 4096 bf16 = 128KB
  const int wave = threadIdx.x >> 5;
  const int lane = threadIdx.x & 31;
  const int m0 = blockIdx.x * 16;
  const int n0 = (blockIdx.y * 8 + wave) * 16;
  const unsigned short* Asrc = X + (size_t)m0 * XDIM;
  const unsigned short* Bbase = Wl_bf + (size_t)n0 * XDIM;  // W_l[n,k] row-major == B columns

  {
    unsigned lbase = (unsigned)(size_t)s_a;
    for (int t = threadIdx.x; t < (16 * XDIM) / 8; t += 256)
      async_g2l_b128(lbase + (unsigned)t * 16u, Asrc + (size_t)t * 8);
    wait_asynccnt0();
    __syncthreads();        // cross-wave visibility of the staged tile
  }

  v8f acc = {};
#pragma unroll 4
  for (int k0 = 0; k0 < XDIM; k0 += 32) {
    __builtin_prefetch(Bbase + (size_t)(lane & 15) * XDIM + k0 + 256, 0, 1);  // global_prefetch_b8
    v16bf a  = load_a(s_a, XDIM, k0, lane);                 // ds_load
    v16bf bm = load_b(Bbase, XDIM, k0, lane);
    acc = __builtin_amdgcn_wmma_f32_16x16x32_bf16(false, a, false, bm, (short)0, acc, false, false);
  }
  const int col = n0 + (lane & 15);
  const float bl = b_l[col];
  const int rbase = (lane >> 4) << 3;
#pragma unroll
  for (int v = 0; v < 8; ++v)
    Z[(size_t)(m0 + v + rbase) * DDIM + col] = acc[v] + bl;
}

// ---------------- launcher ----------------

extern "C" void kernel_launch(void* const* d_in, const int* in_sizes, int n_in,
                              void* d_out, int out_size, void* d_ws, size_t ws_size,
                              hipStream_t stream) {
  const float* c    = (const float*)d_in[0];
  const float* q    = (const float*)d_in[1];
  const float* w_cq = (const float*)d_in[2];
  const float* b_cq = (const float*)d_in[3];
  const float* w_c  = (const float*)d_in[4];
  const float* b_c  = (const float*)d_in[5];
  const float* w_q  = (const float*)d_in[6];
  const float* b_q  = (const float*)d_in[7];
  const float* W_l  = (const float*)d_in[8];
  const float* b_l  = (const float*)d_in[9];
  float* Z = (float*)d_out;

  char* ws = (char*)d_ws;
  size_t off = 0;
  auto alloc = [&](size_t bytes) { char* p = ws + off; off += (bytes + 255) & ~(size_t)255; return p; };

  unsigned short* q_bf   = (unsigned short*)alloc((size_t)B_ * QLEN * DDIM * 2);
  unsigned short* q_t_bf = (unsigned short*)alloc((size_t)B_ * DDIM * QLEN * 2);
  unsigned short* cw_bf  = (unsigned short*)alloc((size_t)B_ * CLEN * DDIM * 2);
  unsigned short* Wl_bf  = (unsigned short*)alloc((size_t)DDIM * XDIM * 2);
  unsigned short* X      = (unsigned short*)alloc((size_t)B_ * CLEN * XDIM * 2);
  float* rq   = (float*)alloc((size_t)B_ * QLEN * 4);
  float* rc   = (float*)alloc((size_t)B_ * CLEN * 4);
  float* mrow = (float*)alloc((size_t)B_ * CLEN * 4);
  float* batt = (float*)alloc((size_t)B_ * CLEN * 4);
  float* q2c  = (float*)alloc((size_t)B_ * DDIM * 4);

  prep_q_kernel<<<B_ * QLEN, 256, 0, stream>>>(q, w_q, b_q, q_bf, q_t_bf, rq);
  prep_c_kernel<<<B_ * CLEN, 256, 0, stream>>>(c, w_cq, w_c, b_c, cw_bf, rc);
  {
    int n = DDIM * XDIM;
    cvt_bf16_kernel<<<(n + 255) / 256, 256, 0, stream>>>(W_l, Wl_bf, n);
  }
  {
    dim3 grid(CLEN / 32, B_);
    size_t lds = (size_t)2 * 16 * DDIM * 2      // A tiles / probs
               + (size_t)2 * 16 * QLEN * 4;     // f32 scores
    attn_kernel<<<grid, 64, lds, stream>>>(cw_bf, q_bf, q_t_bf, rc, rq, b_cq, c, mrow, X);
  }
  batt_kernel<<<B_, 256, 0, stream>>>(mrow, batt);
  {
    dim3 grid(DDIM / 256, B_);
    q2c_kernel<<<grid, 256, 0, stream>>>(batt, c, q2c);
  }
  fillx_kernel<<<B_ * CLEN, 256, 0, stream>>>(c, q2c, X);
  {
    dim3 grid((B_ * CLEN) / 16, DDIM / 128);
    size_t lds = (size_t)16 * XDIM * 2;         // 128KB A tile
    zgemm_kernel<<<grid, 256, lds, stream>>>(X, Wl_bf, b_l, Z);
  }
}
